// PLFConstruct_40415642255578
// MI455X (gfx1250) — matrix operand order repaired and checked by reading, not compiled, not run on hardware
//
#include <hip/hip_runtime.h>
#include <hip/hip_bf16.h>

typedef __attribute__((ext_vector_type(16))) _Float16 v16h;
typedef __attribute__((ext_vector_type(8)))  float    v8f;

#define H_IN 256
#define W_IN 256
#define HH   128
#define WW   128
#define NCH  64
#define KDIM 36
#define KPAD 64

// Shared memory layout (48 KB):
//   region0 (32768 B): union { float in_stage[24][256] (24576B used) ; float out_stage[64][128] }
//   region1 (16384 B): _Float16 psv[8 waves][16 pixels][64 K]
__global__ __launch_bounds__(256) void plf_fused_kernel(const float* __restrict__ in_x,
                                                        const float* __restrict__ w_dd,
                                                        float* __restrict__ out)
{
    __shared__ __align__(16) unsigned char smem[32768 + 16384];
    float*    in_stage  = reinterpret_cast<float*>(smem);            // [24][256]
    float*    out_stage = reinterpret_cast<float*>(smem);            // [64][128] (aliased, barrier-separated)
    _Float16* psv       = reinterpret_cast<_Float16*>(smem + 32768); // [8][16][64]

    const int tid  = threadIdx.x;
    const int wave = tid >> 5;
    const int lane = tid & 31;

    const int y  = blockIdx.x;   // output row, 0..127
    const int a  = blockIdx.y;   // angular index, 0..24
    const int ax = a % 5, ay = a / 5;
    const float dxa = (float)(2 - ax);
    const float dya = (float)(2 - ay);

    // ---- Phase 1: stage in_x rows [2y-10, 2y+13] (clamped) into LDS ----
    {
        const int rowbase = 2 * y - 10;
        // 24 rows * 64 float4 = 1536 float4; 256 threads -> 6 iterations
        for (int it = 0; it < 6; ++it) {
            int idx = it * 256 + tid;       // 0..1535
            int i   = idx >> 6;             // row slot 0..23
            int c4  = idx & 63;             // float4 column
            int r   = rowbase + i;
            r = r < 0 ? 0 : (r > H_IN - 1 ? H_IN - 1 : r);
            float4 v = reinterpret_cast<const float4*>(in_x + (size_t)r * W_IN)[c4];
            reinterpret_cast<float4*>(in_stage + i * W_IN)[c4] = v;
        }
    }
    __syncthreads();

    // ---- Phase 2: compute psv (re-normalized backward bilinear sampling), f16, K padded to 64 ----
    {
        const int x0 = wave * 16;
        _Float16* mypsv = psv + wave * (16 * KPAD);
        for (int it = 0; it < 32; ++it) {
            int idx = it * 32 + lane;       // 0..1023
            int p = idx >> 6;               // pixel within wave, 0..15
            int k = idx & 63;               // psv channel (padded), 0..63
            float res = 0.f;
            if (k < KDIM) {
                int s  = k / 9;             // sub-image 0..3
                int d  = k - s * 9;         // disparity 0..8
                int sy = s >> 1, sx = s & 1;
                // backward-sample offset m = -flow = (4-d)*0.5 * dir
                float mx = (float)(4 - d) * 0.5f * dxa;
                float my = (float)(4 - d) * 0.5f * dya;
                float fox = floorf(mx), foy = floorf(my);
                float fu = mx - fox, fv = my - foy;
                int oxi = (int)fox, oyi = (int)foy;
                int x   = x0 + p;
                int xt0 = x + oxi, xt1 = xt0 + 1;
                int yt0 = y + oyi, yt1 = yt0 + 1;
                float w00 = (1.f - fu) * (1.f - fv);
                float w10 = fu * (1.f - fv);
                float w01 = (1.f - fu) * fv;
                float w11 = fu * fv;
                // validity on the 128x128 sub-image grid (zero weight outside)
                w00 = ((xt0 >= 0) & (xt0 < WW) & (yt0 >= 0) & (yt0 < HH)) ? w00 : 0.f;
                w10 = ((xt1 >= 0) & (xt1 < WW) & (yt0 >= 0) & (yt0 < HH)) ? w10 : 0.f;
                w01 = ((xt0 >= 0) & (xt0 < WW) & (yt1 >= 0) & (yt1 < HH)) ? w01 : 0.f;
                w11 = ((xt1 >= 0) & (xt1 < WW) & (yt1 >= 0) & (yt1 < HH)) ? w11 : 0.f;
                // clamped LDS reads (zero-weight taps read finite data, contribute 0)
                int cx0 = xt0 < 0 ? 0 : (xt0 > WW - 1 ? WW - 1 : xt0);
                int cx1 = xt1 < 0 ? 0 : (xt1 > WW - 1 ? WW - 1 : xt1);
                int r0  = 2 * (yt0 - y + 5) + sy;   // row slot, always in [0,23]
                int r1  = 2 * (yt1 - y + 5) + sy;
                float t00 = in_stage[r0 * W_IN + 2 * cx0 + sx];
                float t10 = in_stage[r0 * W_IN + 2 * cx1 + sx];
                float t01 = in_stage[r1 * W_IN + 2 * cx0 + sx];
                float t11 = in_stage[r1 * W_IN + 2 * cx1 + sx];
                float val = w00 * t00 + w10 * t10 + w01 * t01 + w11 * t11;
                float ws  = w00 + w10 + w01 + w11;
                res = (ws > 0.f) ? val / ws : 0.f;
            }
            mypsv[p * KPAD + k] = (_Float16)res;
        }
    }
    __syncthreads();

    // ---- Phase 3: build B fragments (w_dd, 32x16 16-bit B layout) ----
    v16h bfrag[4][2];
    {
        const int n     = lane & 15;
        const int khalf = (lane & 16) ? 16 : 0;  // lanes 0-15: K 0..15 ; lanes 16-31: K 16..31
#pragma unroll
        for (int nt = 0; nt < 4; ++nt) {
            const int o = nt * 16 + n;
#pragma unroll
            for (int kc = 0; kc < 2; ++kc) {
                const int kb = kc * 32 + khalf;
#pragma unroll
                for (int j = 0; j < 16; ++j) {
                    int K = kb + j;
                    bfrag[nt][kc][j] = (K < KDIM) ? (_Float16)w_dd[o * KDIM + K] : (_Float16)0.f;
                }
            }
        }
    }

    // ---- Phase 4: build A fragments (16x32 16-bit A layout) ----
    v16h afrag[2];
    {
        const int m = lane & 15;
        const _Float16* mypsv = psv + wave * (16 * KPAD) + m * KPAD;
        const int koff = (lane & 16) ? 16 : 0;   // lanes 0-15: K 0..15 ; 16-31: K 16..31
#pragma unroll
        for (int kc = 0; kc < 2; ++kc) {
            const int base = kc * 32 + koff;
#pragma unroll
            for (int j = 0; j < 8; ++j) {
                afrag[kc][j]     = mypsv[base + j];
                afrag[kc][j + 8] = mypsv[base + 8 + j];
            }
        }
    }

    // ---- Phase 5: WMMA  C[16 pix][64 ch] = A[16x64] x B[64x64]  (K padded) ----
    v8f cfrag[4];
#pragma unroll
    for (int nt = 0; nt < 4; ++nt) {
        v8f c = {};
        c = __builtin_amdgcn_wmma_f32_16x16x32_f16(false, afrag[0], false, bfrag[nt][0],
                                                   (short)0, c, false, false);
        c = __builtin_amdgcn_wmma_f32_16x16x32_f16(false, afrag[1], false, bfrag[nt][1],
                                                   (short)0, c, false, false);
        cfrag[nt] = c;
    }

    // ---- Phase 6: scatter C into out_stage (16x16 f32 C/D layout) ----
    {
        const int n     = lane & 15;
        const int mbase = (lane & 16) ? 8 : 0;
#pragma unroll
        for (int nt = 0; nt < 4; ++nt) {
            const int o = nt * 16 + n;
#pragma unroll
            for (int i = 0; i < 8; ++i) {
                out_stage[o * WW + wave * 16 + mbase + i] = cfrag[nt][i];
            }
        }
    }
    __syncthreads();

    // ---- Phase 7: coalesced store out[a][o][y][0..127] ----
    {
        float* dst = out + (size_t)a * NCH * HH * WW + (size_t)y * WW;
        // 64*128 floats = 2048 float4; 256 threads -> 8 iterations
        for (int it = 0; it < 8; ++it) {
            int idx = it * 256 + tid;   // float4 index, 0..2047
            int o   = idx >> 5;         // 32 float4 per channel row
            int c4  = idx & 31;
            float4 v = reinterpret_cast<float4*>(out_stage + o * WW)[c4];
            reinterpret_cast<float4*>(dst + (size_t)o * HH * WW)[c4] = v;
        }
    }
}

extern "C" void kernel_launch(void* const* d_in, const int* in_sizes, int n_in,
                              void* d_out, int out_size, void* d_ws, size_t ws_size,
                              hipStream_t stream) {
    const float* in_x = (const float*)d_in[0];   // 1x1x256x256 f32
    const float* w_dd = (const float*)d_in[1];   // 64x36 f32
    float* out = (float*)d_out;                  // 25x64x128x128 f32
    (void)in_sizes; (void)n_in; (void)out_size; (void)d_ws; (void)ws_size;
    dim3 grid(HH, 25);
    plf_fused_kernel<<<grid, 256, 0, stream>>>(in_x, w_dd, out);
}
